// MobileNet_v1_37572373905425
// MI455X (gfx1250) — matrix operand order repaired and checked by reading, not compiled, not run on hardware
//
#include <hip/hip_runtime.h>

// MobileNetV1 forward, B=32. Depthwise/stem = f32 VALU (bandwidth-bound).
// Pointwise 1x1 convs = bf16 WMMA GEMMs fed from fragment-ready swizzled bf16
// buffers, staged through LDS with CDNA5 async copies (ASYNCcnt) and
// double-buffering so the WMMA loop overlaps global traffic with compute.
// Input order: 0:x 1:conv1_w 2:conv1_s 3:conv1_b, 13x{dw_w,dw_s,dw_b,pw_w,pw_s,pw_b},
//              82:fc_w 83:fc_b

#define B_ 32

typedef __attribute__((ext_vector_type(16))) __bf16 v16bf;
typedef __attribute__((ext_vector_type(8)))  float  v8f;
typedef int v4i_ __attribute__((vector_size(16)));   // matches builtin param type

// ---- CDNA5 async global->LDS copy helpers (guarded; asm fallback) ----------
__device__ __forceinline__ void async_copy16(const __bf16* g, __bf16* l) {
#if defined(__gfx1250__)
#if __has_builtin(__builtin_amdgcn_global_load_async_to_lds_b128)
  __builtin_amdgcn_global_load_async_to_lds_b128(
      (__attribute__((address_space(1))) v4i_*)g,
      (__attribute__((address_space(3))) v4i_*)l, 0, 0);
#else
  unsigned lds_off =
      (unsigned)(__UINTPTR_TYPE__)(__attribute__((address_space(3))) void*)l;
  asm volatile("global_load_async_to_lds_b128 %0, %1, off"
               :: "v"(lds_off), "v"((unsigned long long)(__UINTPTR_TYPE__)g)
               : "memory");
#endif
#else
  *(int4*)l = *(const int4*)g;   // non-gfx1250 / host parse fallback
#endif
}

__device__ __forceinline__ void wait_async_all() {
#if defined(__gfx1250__)
#if __has_builtin(__builtin_amdgcn_s_wait_asynccnt)
  __builtin_amdgcn_s_wait_asynccnt(0);
#else
  asm volatile("s_wait_asynccnt 0x0" ::: "memory");
#endif
#endif
}

__global__ void zero_kernel(float* __restrict__ p, int n) {
  int i = blockIdx.x * blockDim.x + threadIdx.x;
  if (i < n) p[i] = 0.0f;
}

// ---------------- stem: conv3x3 s2 p1 (3->32) + BN + ReLU + per-(b,c) max ----
__global__ void stem_kernel(const float* __restrict__ x, const float* __restrict__ w,
                            const float* __restrict__ s, const float* __restrict__ bias,
                            float* __restrict__ out, float* __restrict__ outmax) {
  const int HO = 112, WO = 112, HI = 224, WI = 224, CO = 32;
  int idx = blockIdx.x * blockDim.x + threadIdx.x;
  int total = B_ * CO * HO * WO;
  if (idx >= total) return;
  int wo = idx % WO; int t = idx / WO;
  int ho = t % HO;   t /= HO;
  int c  = t % CO;   int b = t / CO;
  float acc = 0.f;
  int hb = ho * 2 - 1, wb = wo * 2 - 1;
  for (int kc = 0; kc < 3; ++kc) {
    const float* ip = x + ((size_t)b * 3 + kc) * HI * WI;
    const float* wp = w + (c * 3 + kc) * 9;
#pragma unroll
    for (int r = 0; r < 3; ++r) {
      int hi = hb + r;
      if (hi < 0 || hi >= HI) continue;
#pragma unroll
      for (int q = 0; q < 3; ++q) {
        int wi = wb + q;
        if (wi < 0 || wi >= WI) continue;
        acc += ip[hi * WI + wi] * wp[r * 3 + q];
      }
    }
  }
  float v = acc * s[c] + bias[c];
  v = v > 0.f ? v : 0.f;
  out[idx] = v;
  atomicMax((unsigned int*)&outmax[b * CO + c], __float_as_uint(v));
}

// ---------------- depthwise 3x3 + BN + ReLU, input-channel mask, out max -----
__global__ void dw_kernel(const float* __restrict__ in,  const float* __restrict__ w,
                          const float* __restrict__ s,   const float* __restrict__ bias,
                          const float* __restrict__ inmax, float in_thresh,
                          float* __restrict__ out, float* __restrict__ outmax,
                          int C, int Hin, int Win, int Hout, int Wout, int stride) {
  int idx = blockIdx.x * blockDim.x + threadIdx.x;
  int total = B_ * C * Hout * Wout;
  if (idx >= total) return;
  int wo = idx % Wout; int t = idx / Wout;
  int ho = t % Hout;   t /= Hout;
  int c  = t % C;      int b = t / C;
  float mask = (inmax[b * C + c] >= in_thresh) ? 1.0f : 0.0f;
  const float* ip = in + (size_t)(b * C + c) * Hin * Win;
  const float* wp = w + c * 9;
  float acc = 0.f;
  int hb = ho * stride - 1, wb = wo * stride - 1;
#pragma unroll
  for (int r = 0; r < 3; ++r) {
    int hi = hb + r;
    if (hi < 0 || hi >= Hin) continue;
#pragma unroll
    for (int q = 0; q < 3; ++q) {
      int wi = wb + q;
      if (wi < 0 || wi >= Win) continue;
      acc += ip[hi * Win + wi] * wp[r * 3 + q];
    }
  }
  acc *= mask;                       // linear conv: masking sum == masking input
  float v = acc * s[c] + bias[c];
  v = v > 0.f ? v : 0.f;
  out[idx] = v;
  atomicMax((unsigned int*)&outmax[b * C + c], __float_as_uint(v));
}

// ---------------- repack weights -> A-fragment-ready bf16 --------------------
// Layout: for m, k-tile t, half h: 16 contiguous bf16 at ((m*T + t)*2 + h)*16
//   j=0..15 -> k = t*32 + h*16 + j
__global__ void pack_w_kernel(const float* __restrict__ w, __bf16* __restrict__ bw,
                              int CO, int K) {
  int T = K >> 5;
  int idx = blockIdx.x * blockDim.x + threadIdx.x;
  int total = CO * T * 2;
  if (idx >= total) return;
  int h = idx & 1;
  int t = (idx >> 1) % T;
  int m = (idx >> 1) / T;
  const float* wr = w + (size_t)m * K + t * 32 + h * 16;
  v16bf v;
#pragma unroll
  for (int j = 0; j < 8; ++j) {
    v[j]     = (__bf16)wr[j];
    v[j + 8] = (__bf16)wr[8 + j];
  }
  *(v16bf*)(bw + (size_t)idx * 16) = v;
}

// ---------------- repack activations -> B-fragment-ready bf16 ----------------
// Applies channel-cut mask, converts to bf16, zero-pads n in [N, Npad).
// Layout: (((b*T + t)*2 + h)*Npad + n)*16 ; j -> k = t*32 + h*16 + j
__global__ void pack_act_kernel(const float* __restrict__ in, const float* __restrict__ inmax,
                                float thresh, __bf16* __restrict__ bact,
                                int K, int N, int Npad) {
  int T = K >> 5;
  int idx = blockIdx.x * blockDim.x + threadIdx.x;
  int total = B_ * T * 2 * Npad;
  if (idx >= total) return;
  int n  = idx % Npad;
  int t2 = idx / Npad;             // (b*T + t)*2 + h
  int h  = t2 & 1;
  int t  = (t2 >> 1) % T;
  int b  = (t2 >> 1) / T;
  float nv = (n < N) ? 1.0f : 0.0f;
  int nc = (n < N) ? n : 0;
  int kbase = t * 32 + h * 16;
  const float* mrow = inmax + (size_t)b * K + kbase;
  const float* irow = in + ((size_t)b * K + kbase) * N + nc;
  v16bf v;
#pragma unroll
  for (int j = 0; j < 16; ++j) {
    float m = (mrow[j] >= thresh) ? 1.0f : 0.0f;
    v[j] = (__bf16)(irow[(size_t)j * N] * m * nv);
  }
  *(v16bf*)(bact + (size_t)idx * 16) = v;
}

// ---------------- pointwise 1x1 as bf16 WMMA GEMM ----------------------------
// Block = 256 threads = 8 waves, output tile 64(co) x 64(n) per K-sweep.
// Per K-step: block stages A (64x32 bf16, 4KB) and B (32x64 bf16, 4KB) into
// LDS via async copies (2 x b128 per thread), double-buffered; waves read
// fragments with ds_load_b128 and issue 2 WMMAs (A reused across 2 N-tiles).
__global__ __launch_bounds__(256)
void pw_wmma_kernel(const __bf16* __restrict__ bact,  // swizzled activations
                    const __bf16* __restrict__ bw,    // swizzled weights
                    const float* __restrict__ s, const float* __restrict__ bias,
                    float* __restrict__ out,          // [B,CO,N] f32
                    float* __restrict__ outmax,       // [B,CO]
                    int K, int CO, int N, int Npad) {
  const int T = K >> 5;
  const int tid  = threadIdx.x;
  const int lane = tid & 31;
  const int wave = tid >> 5;
  const int b  = blockIdx.z;
  const int wm = wave & 3, wn = wave >> 2;
  const int co0g = blockIdx.y * 64;
  const int n0g  = blockIdx.x * 64;
  const int lo = lane & 15;
  const int hi = lane >> 4;

  __shared__ __align__(16) __bf16 smA[2 * 2048];   // [buf][m(64)][h(2)][16]
  __shared__ __align__(16) __bf16 smB[2 * 2048];   // [buf][h(2)][n(64)][16]

  // --- per-thread 16B copy chunk assignment (2 chunks: one A, one B) ---
  // A: q=tid -> m=q>>2, h=(q>>1)&1, s8=q&1
  const int am = tid >> 2, ah = (tid >> 1) & 1, as8 = tid & 1;
  const __bf16* gA = bw + (size_t)(co0g + am) * T * 32 + ah * 16 + as8 * 8;
  const int lAoff = am * 32 + ah * 16 + as8 * 8;
  // B: h=q>>7, n_l=(q>>1)&63, s8=q&1
  const int bh = tid >> 7, bn = (tid >> 1) & 63, bs8 = tid & 1;
  const __bf16* gB = bact + ((size_t)b * T * 2 + bh) * Npad * 16
                          + (size_t)(n0g + bn) * 16 + bs8 * 8;
  const size_t bstride = (size_t)2 * Npad * 16;
  const int lBoff = bh * 1024 + bn * 16 + bs8 * 8;

  // --- fragment read offsets (elements) ---
  const int raoff = (wm * 16 + lo) * 32 + hi * 16;
  const int rboff = hi * 1024 + (wn * 32 + lo) * 16;

  // prologue: stage K-tile 0 into buffer 0
  async_copy16(gA, &smA[lAoff]);         gA += 32;
  async_copy16(gB, &smB[lBoff]);         gB += bstride;

  v8f acc0 = {}, acc1 = {};
  for (int t = 0; t < T; ++t) {
    wait_async_all();        // buffer (t&1) fully written (batches complete in order)
    __syncthreads();         // all waves' reads of buffer ((t+1)&1) from step t-1 done
    if (t + 1 < T) {         // prefetch next K-tile into the other buffer
      int nb = (t + 1) & 1;
      async_copy16(gA, &smA[nb * 2048 + lAoff]);  gA += 32;
      async_copy16(gB, &smB[nb * 2048 + lBoff]);  gB += bstride;
    }
    const int cb = (t & 1) * 2048;
    v16bf a  = *(const v16bf*)&smA[cb + raoff];
    v16bf b0 = *(const v16bf*)&smB[cb + rboff];
    v16bf b1 = *(const v16bf*)&smB[cb + rboff + 256];
    acc0 = __builtin_amdgcn_wmma_f32_16x16x32_bf16(
        false, a, false, b0, (short)0, acc0, false, false);
    acc1 = __builtin_amdgcn_wmma_f32_16x16x32_bf16(
        false, a, false, b1, (short)0, acc1, false, false);
  }

  // D layout: VGPR r -> M = r + 8*hi, N-col = lo
  const int co0 = co0g + wm * 16;
  const int n_a = n0g + wn * 32 + lo, n_b = n_a + 16;
#pragma unroll
  for (int r = 0; r < 8; ++r) {
    int co = co0 + r + 8 * hi;
    float sc = s[co], bi = bias[co];
    float va = acc0[r] * sc + bi; va = va > 0.f ? va : 0.f;
    float vb = acc1[r] * sc + bi; vb = vb > 0.f ? vb : 0.f;
    size_t rowoff = ((size_t)b * CO + co) * N;
    unsigned int* mx = (unsigned int*)&outmax[b * CO + co];
    if (n_a < N) { out[rowoff + n_a] = va; atomicMax(mx, __float_as_uint(va)); }
    if (n_b < N) { out[rowoff + n_b] = vb; atomicMax(mx, __float_as_uint(vb)); }
  }
}

// ---------------- global average pool (masked) + FC --------------------------
__global__ void pool_kernel(const float* __restrict__ in, const float* __restrict__ inmax,
                            float thresh, float* __restrict__ out, int HW, int C) {
  int idx = blockIdx.x * blockDim.x + threadIdx.x;
  if (idx >= B_ * C) return;
  float mask = (inmax[idx] >= thresh) ? 1.0f : 0.0f;
  const float* p = in + (size_t)idx * HW;
  float acc = 0.f;
  for (int i = 0; i < HW; ++i) acc += p[i];
  out[idx] = mask * acc / (float)HW;
}

__global__ void fc_kernel(const float* __restrict__ pooled, const float* __restrict__ w,
                          const float* __restrict__ bias, float* __restrict__ out,
                          int C, int NC) {
  int idx = blockIdx.x * blockDim.x + threadIdx.x;
  if (idx >= B_ * NC) return;
  int j = idx % NC, b = idx / NC;
  float acc = bias[j];
  const float* pr = pooled + (size_t)b * C;
  for (int c = 0; c < C; ++c) acc += pr[c] * w[(size_t)c * NC + j];
  out[idx] = acc;
}

// ---------------------------------------------------------------------------
extern "C" void kernel_launch(void* const* d_in, const int* in_sizes, int n_in,
                              void* d_out, int out_size, void* d_ws, size_t ws_size,
                              hipStream_t stream) {
  (void)in_sizes; (void)n_in; (void)out_size; (void)ws_size;
  const float* x   = (const float*)d_in[0];
  const float* c1w = (const float*)d_in[1];
  const float* c1s = (const float*)d_in[2];
  const float* c1b = (const float*)d_in[3];
  const float* fcw = (const float*)d_in[82];
  const float* fcb = (const float*)d_in[83];

  static const int cfg_st[13] = {1,2,1,2,1,2,1,1,1,1,1,2,1};
  static const int cfg_ci[13] = {32,64,128,128,256,256,512,512,512,512,512,512,1024};
  static const int cfg_co[13] = {64,128,128,256,256,512,512,512,512,512,512,1024,1024};

  // Workspace layout (all offsets in floats, 32B-aligned):
  float* buf0   = (float*)d_ws;                    // pw/stem f32 acts (25,690,112)
  float* buf1   = buf0 + (size_t)25690112;         // dw f32 acts      (12,845,056)
  float* max0   = buf1 + (size_t)12845056;         // per-(b,c) max, pw/stem
  float* max1   = max0 + (size_t)(32 * 1024);      // per-(b,c) max, dw
  float* pooled = max1 + (size_t)(32 * 1024);      // [32,1024]
  __bf16* bact  = (__bf16*)(pooled + (size_t)(32 * 1024));  // 12,845,056 bf16 max
  __bf16* bw    = (__bf16*)((float*)bact + (size_t)(12845056 / 2)); // 1,048,576 bf16

  const float PW_T = 1e-3f, DW_T = 4.0f;

  // stem -> buf0 / max0
  zero_kernel<<<(B_ * 32 + 255) / 256, 256, 0, stream>>>(max0, B_ * 32);
  {
    int total = B_ * 32 * 112 * 112;
    stem_kernel<<<(total + 255) / 256, 256, 0, stream>>>(x, c1w, c1s, c1b, buf0, max0);
  }

  int H = 112;
  for (int i = 0; i < 13; ++i) {
    int st = cfg_st[i], ci = cfg_ci[i], co = cfg_co[i];
    int Hin = H, Hout = (st == 2) ? H / 2 : H;
    const float* dww = (const float*)d_in[4 + i * 6 + 0];
    const float* dws = (const float*)d_in[4 + i * 6 + 1];
    const float* dwb = (const float*)d_in[4 + i * 6 + 2];
    const float* pww = (const float*)d_in[4 + i * 6 + 3];
    const float* pws = (const float*)d_in[4 + i * 6 + 4];
    const float* pwb = (const float*)d_in[4 + i * 6 + 5];

    // depthwise: buf0 (masked by max0 @ 1e-3) -> buf1 / max1
    zero_kernel<<<(B_ * ci + 255) / 256, 256, 0, stream>>>(max1, B_ * ci);
    {
      int total = B_ * ci * Hout * Hout;
      dw_kernel<<<(total + 255) / 256, 256, 0, stream>>>(
          buf0, dww, dws, dwb, max0, PW_T, buf1, max1,
          ci, Hin, Hin, Hout, Hout, st);
    }

    int N = Hout * Hout;
    int Npad = ((N + 63) / 64) * 64;
    int T = ci / 32;

    // repack weights and activations into fragment-ready bf16
    {
      int total = co * T * 2;
      pack_w_kernel<<<(total + 255) / 256, 256, 0, stream>>>(pww, bw, co, ci);
    }
    {
      int total = B_ * T * 2 * Npad;
      pack_act_kernel<<<(total + 255) / 256, 256, 0, stream>>>(
          buf1, max1, DW_T, bact, ci, N, Npad);
    }

    // pointwise WMMA GEMM -> buf0 / max0
    zero_kernel<<<(B_ * co + 255) / 256, 256, 0, stream>>>(max0, B_ * co);
    {
      dim3 g(Npad / 64, co / 64, B_);
      pw_wmma_kernel<<<g, 256, 0, stream>>>(
          bact, bw, pws, pwb, buf0, max0, ci, co, N, Npad);
    }
    H = Hout;
  }

  // masked global average pool (7x7) + classifier
  pool_kernel<<<(B_ * 1024 + 255) / 256, 256, 0, stream>>>(buf0, max0, PW_T, pooled, 49, 1024);
  fc_kernel<<<(B_ * 1000 + 255) / 256, 256, 0, stream>>>(pooled, fcw, fcb, (float*)d_out, 1024, 1000);
}
